// SASRec_80839874445932
// MI455X (gfx1250) — matrix operand order repaired and checked by reading, not compiled, not run on hardware
//
#include <hip/hip_runtime.h>

// ---------------------------------------------------------------------------
// SASRec forward + loss for MI455X (gfx1250, wave32, WMMA bf16).
// All GEMMs use v_wmma_f32_16x16x32_bf16 (bf16 inputs, f32 accumulate).
// Q/K/V/CTX live in a 224-rows-per-batch padded layout so all attention
// operand loads are guard-free (vectorizable b128); masking is done on the
// scores with VALU selects. Gram kernels process whole 32-row chunks only
// (f32 tail kernel handles the remainder). All reductions are fixed-order.
// ---------------------------------------------------------------------------

typedef __attribute__((ext_vector_type(16))) __bf16 v16bf;
typedef __attribute__((ext_vector_type(8)))  float  v8f;

static constexpr int BATCH = 512;
static constexpr int SEQ   = 200;
static constexpr int SEQP  = 224;           // padded rows/batch (14 tiles worth)
static constexpr int DIM   = 64;
static constexpr int TOK   = BATCH * SEQ;   // 102400
static constexpr int NIT1  = 100001;        // NITEMS + 1

__device__ __forceinline__ v8f wmma_bf16(v16bf a, v16bf b, v8f c) {
  return __builtin_amdgcn_wmma_f32_16x16x32_bf16(false, a, false, b, (short)0, c,
                                                 false, false);
}

// A-matrix (16x32 bf16) element->K mapping per CDNA5 ISA 7.12.2.
__device__ __forceinline__ int kA(int i, int h) {
  return ((i & 8) ? 16 : 0) + h * 8 + ((i >> 1) & 3) * 2 + (i & 1);
}
// B-matrix (32x16 bf16): same element->K mapping as A (table order).
__device__ __forceinline__ int kB(int i, int h) { return ((i & 8) ? 16 : 0) + h * 8 + (i & 7); }

__device__ __forceinline__ int padrow(int t) { return (t / SEQ) * SEQP + (t % SEQ); }

// ---------------------------------------------------------------------------
// Zero the padding rows (s in [200,224)) of the padded bf16 buffers.
// ---------------------------------------------------------------------------
__global__ void __launch_bounds__(256)
padzero_k(__bf16* __restrict__ q16, __bf16* __restrict__ k16,
          __bf16* __restrict__ v16, __bf16* __restrict__ ctx16) {
  int idx = blockIdx.x * 256 + threadIdx.x;  // 512 * 24 * 64 = 786432
  if (idx >= BATCH * (SEQP - SEQ) * DIM) return;
  int b = idx / ((SEQP - SEQ) * DIM), r = idx % ((SEQP - SEQ) * DIM);
  size_t o = (size_t)b * SEQP * DIM + (size_t)SEQ * DIM + r;
  q16[o] = (__bf16)0.0f; k16[o] = (__bf16)0.0f;
  v16[o] = (__bf16)0.0f; ctx16[o] = (__bf16)0.0f;
}

// ---------------------------------------------------------------------------
// Pre-convert all 12 weight matrices (64x64 f32, row-major W[n][k]) to bf16.
// Layout: wb[l*6*4096 + mat*4096 + n*64 + k], mat: 0=Q 1=K 2=V 3=out 4=fc1 5=fc2.
// ---------------------------------------------------------------------------
__global__ void __launch_bounds__(256)
wconv_k(const float* __restrict__ qkv_w, const float* __restrict__ out_w,
        const float* __restrict__ fc1_w, const float* __restrict__ fc2_w,
        __bf16* __restrict__ wb) {
  int idx = blockIdx.x * 256 + threadIdx.x;  // 2*6*4096 = 49152 total
  if (idx >= 49152) return;
  int l = idx / 24576, rem = idx % 24576;
  int mat = rem / 4096, e = rem % 4096;
  const float* src;
  switch (mat) {
    case 0: case 1: case 2: src = qkv_w + l * 12288 + mat * 4096 + e; break;
    case 3:                 src = out_w + l * 4096 + e;               break;
    case 4:                 src = fc1_w + l * 4096 + e;               break;
    default:                src = fc2_w + l * 4096 + e;               break;
  }
  wb[idx] = (__bf16)(*src);
}

// ---------------------------------------------------------------------------
// Embedding gather: x = item_emb[log] * sqrt(D) + pos_emb[pos]
// ---------------------------------------------------------------------------
__global__ void __launch_bounds__(256)
embed_k(const int* __restrict__ logs, const float* __restrict__ item_emb,
        const float* __restrict__ pos_emb, float* __restrict__ xF,
        __bf16* __restrict__ x16) {
  int wave = threadIdx.x >> 5, lane = threadIdx.x & 31;
  int t = blockIdx.x * 8 + wave;
  if (t >= TOK) return;
  int s = t % SEQ;
  int id = logs[t];
  int pos = (id != 0) ? (s + 1) : 0;
  float v0 = item_emb[id * DIM + lane] * 8.0f + pos_emb[pos * DIM + lane];
  float v1 = item_emb[id * DIM + 32 + lane] * 8.0f + pos_emb[pos * DIM + 32 + lane];
  xF[t * DIM + lane] = v0;
  xF[t * DIM + 32 + lane] = v1;
  x16[t * DIM + lane] = (__bf16)v0;
  x16[t * DIM + 32 + lane] = (__bf16)v1;
}

// ---------------------------------------------------------------------------
// LayerNorm over D=64, one wave per token, optional f32/bf16 out (templated).
// ---------------------------------------------------------------------------
template <int OUTF, int OUTB>
__global__ void __launch_bounds__(256)
ln_k(const float* __restrict__ X, const float* __restrict__ w,
     const float* __restrict__ b, float* __restrict__ outF,
     __bf16* __restrict__ outB) {
  int wave = threadIdx.x >> 5, lane = threadIdx.x & 31;
  int t = blockIdx.x * 8 + wave;
  if (t >= TOK) return;
  float x0 = X[t * DIM + lane], x1 = X[t * DIM + 32 + lane];
  float s = x0 + x1;
  for (int m = 16; m >= 1; m >>= 1) s += __shfl_xor(s, m, 32);
  float mean = s * (1.0f / 64.0f);
  float d0 = x0 - mean, d1 = x1 - mean;
  float v = d0 * d0 + d1 * d1;
  for (int m = 16; m >= 1; m >>= 1) v += __shfl_xor(v, m, 32);
  float inv = rsqrtf(v * (1.0f / 64.0f) + 1e-8f);
  float y0 = d0 * inv * w[lane] + b[lane];
  float y1 = d1 * inv * w[32 + lane] + b[32 + lane];
  if (OUTF) { outF[t * DIM + lane] = y0; outF[t * DIM + 32 + lane] = y1; }
  if (OUTB) { outB[t * DIM + lane] = (__bf16)y0; outB[t * DIM + 32 + lane] = (__bf16)y1; }
}

// ---------------------------------------------------------------------------
// OUT[t][n] = act(A16 @ W16^T + bias (+ resid)). 8 waves/block, 128 tok/block.
// PADIN: A is in padded row layout; PADOUT: outB written in padded layout.
// ---------------------------------------------------------------------------
template <int RELU, int HASRES, int OUTF, int OUTB, int PADIN, int PADOUT>
__global__ void __launch_bounds__(256)
gemm_k(const __bf16* __restrict__ A, const __bf16* __restrict__ W16,
       const float* __restrict__ bias, const float* __restrict__ resid,
       float* __restrict__ outF, __bf16* __restrict__ outB) {
  int wave = threadIdx.x >> 5, lane = threadIdx.x & 31;
  int h = lane >> 4, l15 = lane & 15;
  int tbase = blockIdx.x * 128 + wave * 16;

  v16bf a0, a1;
  int t0 = tbase + l15;
  int rowA = PADIN ? padrow(t0) : t0;
#pragma unroll
  for (int i = 0; i < 16; i++) {
    a0[i] = A[rowA * DIM + kA(i, h)];
    a1[i] = A[rowA * DIM + 32 + kA(i, h)];
  }
#pragma unroll
  for (int nt = 0; nt < 4; nt++) {
    int n = nt * 16 + l15;
    v16bf b0, b1;
#pragma unroll
    for (int i = 0; i < 16; i++) {
      b0[i] = W16[n * DIM + kB(i, h)];
      b1[i] = W16[n * DIM + 32 + kB(i, h)];
    }
    v8f c = {};
    c = wmma_bf16(a0, b0, c);
    c = wmma_bf16(a1, b1, c);
    float bs = bias[n];
#pragma unroll
    for (int r = 0; r < 8; r++) {
      int t = tbase + r + 8 * h;
      float v = c[r] + bs;
      if (HASRES) v += resid[t * DIM + n];
      if (RELU) v = v > 0.0f ? v : 0.0f;
      if (OUTF) outF[t * DIM + n] = v;
      if (OUTB) {
        int ro = PADOUT ? padrow(t) : t;
        outB[ro * DIM + n] = (__bf16)v;
      }
    }
  }
}

// ---------------------------------------------------------------------------
// Fused causal attention, flash-style. Grid = BATCH blocks of 8 waves; each
// wave sweeps (qtile, head) pairs. Q/K/V/CTX in padded (224-row) layout ->
// all loads guard-free; masking applied to scores (VALU selects only).
// P re-layout via per-wave LDS + wave-local s_wait_dscnt.
// ---------------------------------------------------------------------------
__global__ void __launch_bounds__(256)
attn_k(const __bf16* __restrict__ Q, const __bf16* __restrict__ K,
       const __bf16* __restrict__ V, __bf16* __restrict__ CTX) {
  __shared__ __bf16 pb[8][16][32];
  int wave = threadIdx.x >> 5, lane = threadIdx.x & 31;
  int h = lane >> 4, l15 = lane & 15;
  int base = blockIdx.x * SEQP;
  const float scale = 0.17677669529663687f;  // 1/sqrt(32)

  for (int pair = wave; pair < 26; pair += 8) {
    int qt = pair % 13, head = pair / 13;
    int dco = head * 32;

    v16bf aq;
    int qrowA = qt * 16 + l15;
#pragma unroll
    for (int i = 0; i < 16; i++)
      aq[i] = Q[(base + qrowA) * DIM + dco + kA(i, h)];

    float mrow[8], lrow[8];
    v8f c0 = {}, c1 = {};
#pragma unroll
    for (int r = 0; r < 8; r++) { mrow[r] = -1e30f; lrow[r] = 0.0f; }

    int nch = qt / 2 + 1;
    for (int ch = 0; ch < nch; ch++) {
      int kb = ch * 32;
      int k0 = kb + l15, k1 = kb + 16 + l15;
      v16bf b0, b1;
#pragma unroll
      for (int i = 0; i < 16; i++) {
        b0[i] = K[(base + k0) * DIM + dco + kB(i, h)];
        b1[i] = K[(base + k1) * DIM + dco + kB(i, h)];
      }
      v8f s0 = {}, s1 = {};
      s0 = wmma_bf16(aq, b0, s0);
      s1 = wmma_bf16(aq, b1, s1);

      float p0[8], p1[8], alpha[8];
#pragma unroll
      for (int r = 0; r < 8; r++) {
        int qr = qt * 16 + r + 8 * h;
        float v0 = s0[r] * scale, v1 = s1[r] * scale;
        // masks (VALU selects; also squash any garbage from padded K rows)
        if (k0 > qr || k0 >= SEQ || qr >= SEQ) v0 = -1e30f;
        if (k1 > qr || k1 >= SEQ || qr >= SEQ) v1 = -1e30f;
        float mx = fmaxf(v0, v1);
        for (int m = 8; m >= 1; m >>= 1) mx = fmaxf(mx, __shfl_xor(mx, m, 32));
        float mnew = fmaxf(mrow[r], mx);
        alpha[r] = __expf(mrow[r] - mnew);
        mrow[r] = mnew;
        p0[r] = __expf(v0 - mnew);
        p1[r] = __expf(v1 - mnew);
        float rs = p0[r] + p1[r];
        for (int m = 8; m >= 1; m >>= 1) rs += __shfl_xor(rs, m, 32);
        lrow[r] = lrow[r] * alpha[r] + rs;
      }
      // Stage P through per-wave LDS: C-fragment layout -> A-fragment layout.
#pragma unroll
      for (int r = 0; r < 8; r++) {
        pb[wave][r + 8 * h][l15] = (__bf16)p0[r];
        pb[wave][r + 8 * h][16 + l15] = (__bf16)p1[r];
      }
      asm volatile("s_wait_dscnt 0" ::: "memory");  // wave-local LDS RAW fence
      v16bf ap;
#pragma unroll
      for (int i = 0; i < 16; i++) ap[i] = pb[wave][l15][kA(i, h)];
      asm volatile("s_wait_dscnt 0" ::: "memory");  // reads done before overwrite

      v16bf bv0, bv1;
#pragma unroll
      for (int i = 0; i < 16; i++) {
        int kt = kb + kB(i, h);  // padded V rows are zero -> p*0 = 0, no NaN
        bv0[i] = V[(base + kt) * DIM + dco + l15];
        bv1[i] = V[(base + kt) * DIM + dco + 16 + l15];
      }
#pragma unroll
      for (int r = 0; r < 8; r++) { c0[r] *= alpha[r]; c1[r] *= alpha[r]; }
      c0 = wmma_bf16(ap, bv0, c0);
      c1 = wmma_bf16(ap, bv1, c1);
    }
#pragma unroll
    for (int r = 0; r < 8; r++) {
      int qr = qt * 16 + r + 8 * h;  // pad rows written but never read
      float inv = 1.0f / lrow[r];
      CTX[(base + qr) * DIM + dco + l15] = (__bf16)(c0[r] * inv);
      CTX[(base + qr) * DIM + dco + 16 + l15] = (__bf16)(c1[r] * inv);
    }
  }
}

// ---------------------------------------------------------------------------
// Gram matrix G = X^T X (64x64): guard-free full 32-row chunks only.
// Each wave holds the full 64x64 in 16 C-fragments; block-serial LDS reduce.
// ---------------------------------------------------------------------------
static constexpr int GRAM_BLOCKS = 64;

__global__ void __launch_bounds__(256)
gram_k(const float* __restrict__ X, int nch, float* __restrict__ partials) {
  __shared__ float acc[4096];
  int wave = threadIdx.x >> 5, lane = threadIdx.x & 31;
  int h = lane >> 4, l15 = lane & 15;

  v8f c[16];
  v8f zero = {};
#pragma unroll
  for (int i = 0; i < 16; i++) c[i] = zero;

  for (int ch = blockIdx.x * 8 + wave; ch < nch; ch += GRAM_BLOCKS * 8) {
    int kb = ch * 32;
    v16bf am[4], bn[4];
#pragma unroll
    for (int mt = 0; mt < 4; mt++) {
#pragma unroll
      for (int i = 0; i < 16; i++) {
        am[mt][i] = (__bf16)X[(kb + kA(i, h)) * DIM + mt * 16 + l15];
        bn[mt][i] = (__bf16)X[(kb + kB(i, h)) * DIM + mt * 16 + l15];
      }
    }
#pragma unroll
    for (int mt = 0; mt < 4; mt++)
#pragma unroll
      for (int nt = 0; nt < 4; nt++)
        c[mt * 4 + nt] = wmma_bf16(am[mt], bn[nt], c[mt * 4 + nt]);
  }

  for (int i = threadIdx.x; i < 4096; i += 256) acc[i] = 0.0f;
  __syncthreads();
  for (int w = 0; w < 8; w++) {  // serial across waves -> deterministic
    if (wave == w) {
#pragma unroll
      for (int mt = 0; mt < 4; mt++)
#pragma unroll
        for (int nt = 0; nt < 4; nt++)
#pragma unroll
          for (int r = 0; r < 8; r++)
            acc[(mt * 16 + r + 8 * h) * 64 + nt * 16 + l15] += c[mt * 4 + nt][r];
    }
    __syncthreads();
  }
  for (int i = threadIdx.x; i < 4096; i += 256)
    partials[(size_t)blockIdx.x * 4096 + i] = acc[i];
}

__global__ void __launch_bounds__(256)
gram_reduce_k(const float* __restrict__ partials, float* __restrict__ G) {
  int i = blockIdx.x * 256 + threadIdx.x;  // 16 blocks -> 4096
  float s = 0.0f;
  for (int bk = 0; bk < GRAM_BLOCKS; bk++) s += partials[(size_t)bk * 4096 + i];
  G[i] = s;
}

// Tail rows [start, start+count): G += sum_r x_r x_r^T  (f32, single block).
__global__ void __launch_bounds__(256)
gram_tail_k(const float* __restrict__ X, int start, int count,
            float* __restrict__ G) {
  for (int idx = threadIdx.x; idx < 4096; idx += 256) {
    int i = idx >> 6, j = idx & 63;
    float s = 0.0f;
    for (int r = 0; r < count; r++)
      s += X[(start + r) * DIM + i] * X[(start + r) * DIM + j];
    G[idx] += s;
  }
}

// left = C * sum_ij FF[ij] * EE[ij] * pw[i] * pw[j]
__global__ void __launch_bounds__(256)
left_k(const float* __restrict__ FF, const float* __restrict__ EE,
       const float* __restrict__ pw, float* __restrict__ out) {
  __shared__ float red[256];
  float s = 0.0f;
  for (int idx = threadIdx.x; idx < 4096; idx += 256) {
    int i = idx >> 6, j = idx & 63;
    s += FF[idx] * EE[idx] * pw[i] * pw[j];
  }
  red[threadIdx.x] = s;
  __syncthreads();
  for (int o = 128; o >= 1; o >>= 1) {
    if (threadIdx.x < o) red[threadIdx.x] += red[threadIdx.x + o];
    __syncthreads();
  }
  if (threadIdx.x == 0) out[0] = 0.001f * red[0];
}

// right partials: ps = sum_d x[t][d]*item_emb[pos[t]][d]*pw[d]; (1-C)ps^2 - 2ps
__global__ void __launch_bounds__(256)
right_k(const float* __restrict__ xF, const float* __restrict__ item_emb,
        const int* __restrict__ pos_seqs, const float* __restrict__ pw,
        float* __restrict__ partials) {
  __shared__ float red[256];
  float s = 0.0f;
  for (int t = blockIdx.x * 256 + threadIdx.x; t < TOK; t += 256 * 256) {
    int id = pos_seqs[t];
    float ps = 0.0f;
#pragma unroll 8
    for (int d = 0; d < DIM; d++)
      ps += xF[t * DIM + d] * item_emb[id * DIM + d] * pw[d];
    s += 0.999f * ps * ps - 2.0f * ps;
  }
  red[threadIdx.x] = s;
  __syncthreads();
  for (int o = 128; o >= 1; o >>= 1) {
    if (threadIdx.x < o) red[threadIdx.x] += red[threadIdx.x + o];
    __syncthreads();
  }
  if (threadIdx.x == 0) partials[blockIdx.x] = red[0];
}

// Frobenius norms: 31 segments (3 global params + 14 tensors x 2 layers).
__global__ void __launch_bounds__(256)
frob_k(const float* item_emb, const float* pos_emb, const float* pred_w,
       const float* ln_w, const float* ln_b, const float* qkv_w, const float* qkv_b,
       const float* out_w, const float* out_b, const float* fc1_w, const float* fc1_b,
       const float* ffln_w, const float* ffln_b, const float* fc2_w, const float* fc2_b,
       const float* ffln2_w, const float* ffln2_b, float* __restrict__ regs) {
  __shared__ float red[256];
  int seg = blockIdx.x;
  const float* p;
  int len;
  if (seg == 0) { p = item_emb; len = NIT1 * DIM; }
  else if (seg == 1) { p = pos_emb; len = (SEQ + 1) * DIM; }
  else if (seg == 2) { p = pred_w; len = DIM; }
  else {
    int k = seg - 3;
    int l = k & 1, tid = k >> 1;
    switch (tid) {
      case 0:  p = ln_w + l * 64;       len = 64;    break;
      case 1:  p = ln_b + l * 64;       len = 64;    break;
      case 2:  p = qkv_w + l * 12288;   len = 12288; break;
      case 3:  p = qkv_b + l * 192;     len = 192;   break;
      case 4:  p = out_w + l * 4096;    len = 4096;  break;
      case 5:  p = out_b + l * 64;      len = 64;    break;
      case 6:  p = fc1_w + l * 4096;    len = 4096;  break;
      case 7:  p = fc1_b + l * 64;      len = 64;    break;
      case 8:  p = ffln_w + l * 64;     len = 64;    break;
      case 9:  p = ffln_b + l * 64;     len = 64;    break;
      case 10: p = fc2_w + l * 4096;    len = 4096;  break;
      case 11: p = fc2_b + l * 64;      len = 64;    break;
      case 12: p = ffln2_w + l * 64;    len = 64;    break;
      default: p = ffln2_b + l * 64;    len = 64;    break;
    }
  }
  float s = 0.0f;
  for (int i = threadIdx.x; i < len; i += 256) { float v = p[i]; s += v * v; }
  red[threadIdx.x] = s;
  __syncthreads();
  for (int o = 128; o >= 1; o >>= 1) {
    if (threadIdx.x < o) red[threadIdx.x] += red[threadIdx.x + o];
    __syncthreads();
  }
  if (threadIdx.x == 0) regs[seg] = sqrtf(red[0]);
}

__global__ void combine_k(const float* __restrict__ leftv,
                          const float* __restrict__ rightP,
                          const float* __restrict__ regs,
                          float* __restrict__ out) {
  if (threadIdx.x == 0) {
    float s = leftv[0];
    for (int i = 0; i < 256; i++) s += rightP[i];
    float rg = 0.0f;
    for (int i = 0; i < 31; i++) rg += regs[i];
    out[0] = s + 0.1f * rg;
  }
}

// ---------------------------------------------------------------------------
// Host-side launch sequence (graph-capture safe: stream-only API).
// ---------------------------------------------------------------------------
extern "C" void kernel_launch(void* const* d_in, const int* in_sizes, int n_in,
                              void* d_out, int out_size, void* d_ws, size_t ws_size,
                              hipStream_t stream) {
  (void)in_sizes; (void)n_in; (void)out_size; (void)ws_size;
  const int*   log_seqs = (const int*)d_in[1];
  const int*   pos_seqs = (const int*)d_in[2];
  const float* item_emb = (const float*)d_in[3];
  const float* pos_emb  = (const float*)d_in[4];
  const float* pred_w   = (const float*)d_in[5];
  const float* ln_w     = (const float*)d_in[6];
  const float* ln_b     = (const float*)d_in[7];
  const float* qkv_w    = (const float*)d_in[8];
  const float* qkv_b    = (const float*)d_in[9];
  const float* out_w    = (const float*)d_in[10];
  const float* out_b    = (const float*)d_in[11];
  const float* fc1_w    = (const float*)d_in[12];
  const float* fc1_b    = (const float*)d_in[13];
  const float* ffln_w   = (const float*)d_in[14];
  const float* ffln_b   = (const float*)d_in[15];
  const float* fc2_w    = (const float*)d_in[16];
  const float* fc2_b    = (const float*)d_in[17];
  const float* ffln2_w  = (const float*)d_in[18];
  const float* ffln2_b  = (const float*)d_in[19];

  // workspace carving (256-byte aligned)
  char* wsb = (char*)d_ws;
  size_t off = 0;
  auto take = [&](size_t bytes) -> char* {
    char* p = wsb + off;
    off += (bytes + 255) & ~(size_t)255;
    return p;
  };
  const size_t PADTOK = (size_t)BATCH * SEQP;  // 114688 padded rows
  float*  xF    = (float*)take((size_t)TOK * DIM * 4);
  float*  qinF  = (float*)take((size_t)TOK * DIM * 4);
  __bf16* x16   = (__bf16*)take((size_t)TOK * DIM * 2);
  __bf16* qin16 = (__bf16*)take((size_t)TOK * DIM * 2);
  __bf16* q16   = (__bf16*)take(PADTOK * DIM * 2);
  __bf16* k16   = (__bf16*)take(PADTOK * DIM * 2);
  __bf16* v16   = (__bf16*)take(PADTOK * DIM * 2);
  __bf16* ctx16 = (__bf16*)take(PADTOK * DIM * 2);
  __bf16* wb16  = (__bf16*)take((size_t)49152 * 2);
  float*  gramP = (float*)take((size_t)GRAM_BLOCKS * 4096 * 4);
  float*  FF    = (float*)take(4096 * 4);
  float*  EE    = (float*)take(4096 * 4);
  float*  leftv = (float*)take(256);
  float*  rightP= (float*)take(256 * 4);
  float*  regs  = (float*)take(256);

  const int tokBlocks = TOK / 8;     // 12800
  const int gemmBlocks = TOK / 128;  // 800
  const int padElems = BATCH * (SEQP - SEQ) * DIM;

  padzero_k<<<(padElems + 255) / 256, 256, 0, stream>>>(q16, k16, v16, ctx16);
  wconv_k<<<192, 256, 0, stream>>>(qkv_w, out_w, fc1_w, fc2_w, wb16);
  embed_k<<<tokBlocks, 256, 0, stream>>>(log_seqs, item_emb, pos_emb, xF, x16);

  for (int l = 0; l < 2; l++) {
    const __bf16* wQ   = wb16 + (size_t)l * 24576;
    const __bf16* wK   = wQ + 4096;
    const __bf16* wV   = wQ + 2 * 4096;
    const __bf16* wOut = wQ + 3 * 4096;
    const __bf16* wF1  = wQ + 4 * 4096;
    const __bf16* wF2  = wQ + 5 * 4096;
    const float* qkvB = qkv_b + (size_t)l * 3 * DIM;

    // q_in = LN(x) (f32 kept for residual, bf16 for Q GEMM)
    ln_k<1, 1><<<tokBlocks, 256, 0, stream>>>(xF, ln_w + l * 64, ln_b + l * 64, qinF, qin16);
    // Q/K/V projections (WMMA), outputs in padded layout
    gemm_k<0, 0, 0, 1, 0, 1><<<gemmBlocks, 256, 0, stream>>>(qin16, wQ, qkvB,       nullptr, nullptr, q16);
    gemm_k<0, 0, 0, 1, 0, 1><<<gemmBlocks, 256, 0, stream>>>(x16,   wK, qkvB + 64,  nullptr, nullptr, k16);
    gemm_k<0, 0, 0, 1, 0, 1><<<gemmBlocks, 256, 0, stream>>>(x16,   wV, qkvB + 128, nullptr, nullptr, v16);
    // fused causal attention
    attn_k<<<BATCH, 256, 0, stream>>>(q16, k16, v16, ctx16);
    // x = q_in + ctx @ out_w^T + out_b (ctx read from padded layout)
    gemm_k<0, 1, 1, 0, 1, 0><<<gemmBlocks, 256, 0, stream>>>(ctx16, wOut, out_b + l * 64, qinF, xF, nullptr);
    // h = relu(LN(x; ffln) @ fc1^T + fc1_b)
    ln_k<0, 1><<<tokBlocks, 256, 0, stream>>>(xF, ffln_w + l * 64, ffln_b + l * 64, nullptr, x16);
    gemm_k<1, 0, 0, 1, 0, 0><<<gemmBlocks, 256, 0, stream>>>(x16, wF1, fc1_b + l * 64, nullptr, nullptr, qin16);
    // x = h @ fc2^T + fc2_b + LN(x; ffln2)
    ln_k<1, 0><<<tokBlocks, 256, 0, stream>>>(xF, ffln2_w + l * 64, ffln2_b + l * 64, qinF, nullptr);
    gemm_k<0, 1, 1, 1, 0, 0><<<gemmBlocks, 256, 0, stream>>>(qin16, wF2, fc2_b + l * 64, qinF, xF, x16);
  }

  // FF = X^T X over tokens (3200 full chunks); EE = E^T E (3125 chunks + 1 tail row)
  gram_k<<<GRAM_BLOCKS, 256, 0, stream>>>(xF, TOK / 32, gramP);
  gram_reduce_k<<<16, 256, 0, stream>>>(gramP, FF);
  gram_k<<<GRAM_BLOCKS, 256, 0, stream>>>(item_emb, NIT1 / 32, gramP);
  gram_reduce_k<<<16, 256, 0, stream>>>(gramP, EE);
  gram_tail_k<<<1, 256, 0, stream>>>(item_emb, (NIT1 / 32) * 32, NIT1 % 32, EE);

  left_k<<<1, 256, 0, stream>>>(FF, EE, pred_w, leftv);
  right_k<<<256, 256, 0, stream>>>(xF, item_emb, pos_seqs, pred_w, rightP);
  frob_k<<<31, 256, 0, stream>>>(item_emb, pos_emb, pred_w, ln_w, ln_b, qkv_w, qkv_b,
                                 out_w, out_b, fc1_w, fc1_b, ffln_w, ffln_b,
                                 fc2_w, fc2_b, ffln2_w, ffln2_b, regs);
  combine_k<<<1, 32, 0, stream>>>(leftv, rightP, regs, (float*)d_out);
}
